// PModel_3633542332420
// MI455X (gfx1250) — compile-verified, matrix operand order checked by reading, not run on hardware
//
#include <hip/hip_runtime.h>
#include <hip/hip_bf16.h>

typedef __attribute__((ext_vector_type(16))) _Float16 v16h;
typedef __attribute__((ext_vector_type(8)))  float    v8f;
typedef int v4i __attribute__((vector_size(16)));

#define AS1 __attribute__((address_space(1)))
#define AS3 __attribute__((address_space(3)))

#if defined(__has_builtin)
# if __has_builtin(__builtin_amdgcn_global_load_async_to_lds_b128)
#  define HAVE_ASYNC_LOAD 1
# endif
# if __has_builtin(__builtin_amdgcn_global_store_async_from_lds_b128)
#  define HAVE_ASYNC_STORE 1
# endif
# if __has_builtin(__builtin_amdgcn_s_wait_asynccnt)
#  define ASYNC_WAIT() __builtin_amdgcn_s_wait_asynccnt(0)
# else
#  define ASYNC_WAIT() asm volatile("s_wait_asynccnt 0x0" ::: "memory")
# endif
#endif

// ---------------- problem constants (match setup_inputs) ----------------
#define B_      2
#define Zc      8
#define Hc      91
#define Wc      180
#define Cc      128
#define HEADS   8
#define HD      16
#define HID     320
#define Lc      (Zc*Hc*Wc)          // 131040
#define TOK     (B_*Lc)             // 262080
#define Hp      (Hc+1)              // 92
#define VOLP    (Zc*Hp*Wc)          // 132480 tokens per batch (padded, rolled)
#define NWIN_B  (VOLP/32)           // 4140 windows per batch
#define NWB     (B_*NWIN_B)         // 8280 windows
#define WIN     32

// ---------------- workspace layout (bytes) ----------------
#define OFF_QKVW   ((size_t)0)                      // 384*128 f16
#define OFF_PROJW  ((size_t)98304)                  // 128*128 f16
#define OFF_CROPW  ((size_t)131072)                 // 2*128*128 f16 [hk][co][ci]
#define OFF_WG     ((size_t)196608)                 // 320*128 f16
#define OFF_WX     ((size_t)278528)                 // 320*128 f16
#define OFF_W2     ((size_t)360448)                 // 128*320 f16
#define OFF_BIAS   ((size_t)442368)                 // 8*32*32 f32
#define OFF_ATTN   ((size_t)475136)                 // NWB*32*128 f16
#define OFF_XRES   (OFF_ATTN + (size_t)NWB*32*128*2) // TOK*128 f32

// ---------------- WMMA fragment helpers (wave32, 16x16x32 f16) ----------------
__device__ __forceinline__ v16h ld_frag(const _Float16* base, int row0, int ldk, int k0) {
    int lane = threadIdx.x & 31;
    const _Float16* p = base + (size_t)(row0 + (lane & 15)) * ldk + k0 + ((lane >> 4) << 3);
    v16h f;
#pragma unroll
    for (int e = 0; e < 8; ++e) { f[e] = p[e]; f[8 + e] = p[16 + e]; }
    return f;
}

// half-K fragment: only K=0..15 valid (hd=16); upper 16 K zeroed.
__device__ __forceinline__ v16h ld_frag_k16(const _Float16* base, int row0, int ldk, int k0) {
    int lane = threadIdx.x & 31;
    const _Float16* p = base + (size_t)(row0 + (lane & 15)) * ldk + k0 + ((lane >> 4) << 3);
    v16h f;
#pragma unroll
    for (int e = 0; e < 8; ++e) { f[e] = p[e]; f[8 + e] = (_Float16)0.0f; }
    return f;
}

// B fragment for P@V: source stored [K tokens][ld], want B[N=d][K=token]
__device__ __forceinline__ v16h ld_fragB_T(const _Float16* base, int ld) {
    int lane = threadIdx.x & 31;
    int n  = lane & 15;
    int kb = (lane >> 4) << 3;
    v16h f;
#pragma unroll
    for (int e = 0; e < 8; ++e) {
        f[e]     = base[(size_t)(kb + e)      * ld + n];
        f[8 + e] = base[(size_t)(16 + kb + e) * ld + n];
    }
    return f;
}

__device__ __forceinline__ v8f wmma_f16(v16h a, v16h b, v8f c) {
    return __builtin_amdgcn_wmma_f32_16x16x32_f16(false, a, false, b, (short)0, c, false, false);
}

// ---------------- prep kernels ----------------
__global__ void k_cast_f16(const float* __restrict__ s, _Float16* __restrict__ d, int n) {
    for (int i = blockIdx.x * blockDim.x + threadIdx.x; i < n; i += gridDim.x * blockDim.x)
        d[i] = (_Float16)s[i];
}

// crop_w (1,2,1,Ci,Co) DHWIO -> f16 [hk][co][ci]
__global__ void k_cast_crop(const float* __restrict__ s, _Float16* __restrict__ d) {
    int i = blockIdx.x * blockDim.x + threadIdx.x;
    if (i >= 2 * Cc * Cc) return;
    int hk = i >> 14, co = (i >> 7) & 127, ci = i & 127;
    d[i] = (_Float16)s[hk * Cc * Cc + ci * Cc + co];
}

// bias_table (1024, heads) -> [head][q][k]
__global__ void k_bias_expand(const float* __restrict__ t, float* __restrict__ d) {
    int i = blockIdx.x * blockDim.x + threadIdx.x;
    if (i >= HEADS * WIN * WIN) return;
    int h = i >> 10, q = (i >> 5) & 31, k = i & 31;
    int zq = q >> 4, hq = (q >> 2) & 3, wq = q & 3;
    int zk = k >> 4, hk = (k >> 2) & 3, wk = k & 3;
    int pos = (zq + 2 * zk) * 112 + (hq + 4 * hk) * 7 + (wq - wk + 3);
    d[i] = t[pos * HEADS + h];
}

// ---------------- kernel 1: fused window attention ----------------
__global__ __launch_bounds__(256) void k_attn(
    const float* __restrict__ x, const float* __restrict__ b_qkv,
    const float* __restrict__ b_proj,
    const _Float16* __restrict__ qkv_w, const _Float16* __restrict__ proj_w,
    const float* __restrict__ bias, _Float16* __restrict__ attn_out)
{
    __shared__ __align__(16) _Float16 xw[WIN * Cc];          // 8 KB (reused as proj out)
    __shared__ __align__(16) _Float16 qkv[WIN * 3 * Cc];     // 24 KB
    __shared__ float    s_s[HEADS * WIN * WIN];              // 32 KB
    __shared__ __align__(16) _Float16 s_p[HEADS * WIN * WIN];// 16 KB
    __shared__ __align__(16) _Float16 cat[WIN * Cc];         // 8 KB
    __shared__ int      s_reg[WIN];

    const int tid  = threadIdx.x;
    const int lane = tid & 31;
    const int wave = tid >> 5;
    const int colL = lane & 15;
    const int rb   = (lane >> 4) << 3;

    const int n  = blockIdx.x;          // window (contiguous 32-token chunk)
    const int b  = n / NWIN_B;
    const int nb = n % NWIN_B;

    // warm the resident weights (speculative prefetch -> global_prefetch_b8)
    __builtin_prefetch(qkv_w + tid * 384, 0, 1);
    __builtin_prefetch(proj_w + tid * 64, 0, 1);

    // ---- phase 1: gather rolled+padded tokens, region codes ----
    {
        int i  = tid >> 3;
        int c0 = (tid & 7) << 4;
        int t  = nb * 32 + i;                       // flat over (Zc,Hp,Wc)
        int wr = t % Wc, hr = (t / Wc) % Hp, zr = t / (Wc * Hp);
        int zs = (zr + Zc - 1) % Zc;
        int hs = (hr + Hp - 2) % Hp;
        int ws = (wr + Wc - 2) % Wc;
        if (hs == 0) {
#pragma unroll
            for (int c = 0; c < 16; ++c) xw[i * Cc + c0 + c] = (_Float16)0.0f;
        } else {
            const float* src = x + ((size_t)b * Lc + (size_t)zs * Hc * Wc + (size_t)(hs - 1) * Wc + ws) * Cc + c0;
#pragma unroll
            for (int c = 0; c < 16; ++c) xw[i * Cc + c0 + c] = (_Float16)src[c];
        }
        if (tid < WIN) {  // region code via CORRECT window partition (mask semantics)
            int hb = (nb / 45) % 23, zb = nb / (45 * 23);
            int wz = tid >> 4, wh = (tid >> 2) & 3;
            int z = zb * 2 + wz, h = hb * 4 + wh;
            int rz = (z < Zc - 2) ? 0 : ((z < Zc - 1) ? 1 : 2);
            int rh = (h < Hp - 4) ? 0 : ((h < Hp - 2) ? 1 : 2);
            s_reg[tid] = rz * 3 + rh;
        }
    }
    __syncthreads();

    // ---- phase 2: QKV GEMM (32x128)@(128x384), wave w -> N tiles w,w+8,w+16 ----
#pragma unroll
    for (int j = 0; j < 3; ++j) {
        int nt = wave + j * 8;
        int n0 = nt * 16;
        v8f a0 = {}, a1 = {};
#pragma unroll
        for (int kt = 0; kt < 4; ++kt) {
            v16h fa0 = ld_frag(xw, 0, Cc, kt * 32);
            v16h fa1 = ld_frag(xw, 16, Cc, kt * 32);
            v16h fb  = ld_frag(qkv_w, n0, Cc, kt * 32);
            a0 = wmma_f16(fa0, fb, a0);
            a1 = wmma_f16(fa1, fb, a1);
        }
        int col = n0 + colL;
        float bq = b_qkv[col];
        float sc = (col < Cc) ? 0.25f : 1.0f;     // q * hd^-0.5
#pragma unroll
        for (int r = 0; r < 8; ++r) {
            qkv[(rb + r) * 384 + col]      = (_Float16)((a0[r] + bq) * sc);
            qkv[(16 + rb + r) * 384 + col] = (_Float16)((a1[r] + bq) * sc);
        }
    }
    __syncthreads();

    // ---- phase 3: per-head attention (wave == head) ----
    {
        const int h = wave;
        float* S = s_s + h * WIN * WIN;
        // S = q k^T  (K = hd = 16, zero-padded to 32)
#pragma unroll
        for (int mt = 0; mt < 2; ++mt)
#pragma unroll
            for (int nt = 0; nt < 2; ++nt) {
                v16h fa = ld_frag_k16(qkv, mt * 16, 384, h * 16);
                v16h fb = ld_frag_k16(qkv, nt * 16, 384, Cc + h * 16);
                v8f acc = {};
                acc = wmma_f16(fa, fb, acc);
                int col = nt * 16 + colL;
#pragma unroll
                for (int r = 0; r < 8; ++r) S[(mt * 16 + rb + r) * WIN + col] = acc[r];
            }
        // softmax (lane = query row) with bias + roll mask
        {
            int q = lane;
            int rq = s_reg[q];
            const float* bh = bias + h * WIN * WIN + q * WIN;
            float vv[WIN];
            float mx = -1e30f;
#pragma unroll
            for (int k = 0; k < WIN; ++k) {
                float v = S[q * WIN + k] + bh[k] + ((rq != s_reg[k]) ? -100.0f : 0.0f);
                vv[k] = v; mx = fmaxf(mx, v);
            }
            float sum = 0.0f;
#pragma unroll
            for (int k = 0; k < WIN; ++k) { vv[k] = __expf(vv[k] - mx); sum += vv[k]; }
            float inv = 1.0f / sum;
            _Float16* P = s_p + h * WIN * WIN + q * WIN;
#pragma unroll
            for (int k = 0; k < WIN; ++k) P[k] = (_Float16)(vv[k] * inv);
        }
        // out = P @ v  (32x32 @ 32x16)
#pragma unroll
        for (int mt = 0; mt < 2; ++mt) {
            v16h fa = ld_frag(s_p + h * WIN * WIN, mt * 16, WIN, 0);
            v16h fb = ld_fragB_T(qkv + 2 * Cc + h * 16, 384);
            v8f acc = {};
            acc = wmma_f16(fa, fb, acc);
            int col = h * 16 + colL;
#pragma unroll
            for (int r = 0; r < 8; ++r) cat[(mt * 16 + rb + r) * Cc + col] = (_Float16)acc[r];
        }
    }
    __syncthreads();

    // ---- phase 4: proj (32x128)@(128x128), staged to LDS, coalesced out ----
    {
        int n0 = wave * 16;
        v8f a0 = {}, a1 = {};
#pragma unroll
        for (int kt = 0; kt < 4; ++kt) {
            v16h fa0 = ld_frag(cat, 0, Cc, kt * 32);
            v16h fa1 = ld_frag(cat, 16, Cc, kt * 32);
            v16h fb  = ld_frag(proj_w, n0, Cc, kt * 32);
            a0 = wmma_f16(fa0, fb, a0);
            a1 = wmma_f16(fa1, fb, a1);
        }
        int col = n0 + colL;
        float bp = b_proj[col];
#pragma unroll
        for (int r = 0; r < 8; ++r) {
            xw[(rb + r) * Cc + col]      = (_Float16)(a0[r] + bp);
            xw[(16 + rb + r) * Cc + col] = (_Float16)(a1[r] + bp);
        }
    }
    __syncthreads();
    {
        _Float16* dst = attn_out + (size_t)n * WIN * Cc;
        int i = tid >> 3, c0 = (tid & 7) << 4;
        int off = i * Cc + c0;                 // 32 bytes per thread, contiguous tile
#ifdef HAVE_ASYNC_STORE
        __builtin_amdgcn_global_store_async_from_lds_b128(
            (AS1 v4i*)(uintptr_t)(dst + off),     (AS3 v4i*)(uintptr_t)&xw[off], 0, 0);
        __builtin_amdgcn_global_store_async_from_lds_b128(
            (AS1 v4i*)(uintptr_t)(dst + off + 8), (AS3 v4i*)(uintptr_t)&xw[off + 8], 0, 0);
        // ASYNCcnt drains at s_endpgm (implicit wait-idle)
#else
        const uint4* s = (const uint4*)&xw[off];
        uint4* d4 = (uint4*)(dst + off);
        d4[0] = s[0]; d4[1] = s[1];
#endif
    }
}

// ---------------- kernel 2: crop conv (GEMM) + LayerNorm + residual ----------------
__global__ __launch_bounds__(256) void k_conv_ln(
    const float* __restrict__ x, const _Float16* __restrict__ attn_out,
    const _Float16* __restrict__ crop_w, const float* __restrict__ crop_b,
    const float* __restrict__ g1, const float* __restrict__ be1,
    float* __restrict__ x_res)
{
    __shared__ __align__(16) _Float16 aw[2][WIN * Cc];   // 16 KB gathered taps
    __shared__ float    cv[WIN * Cc];                    // 16 KB conv result
    __shared__ float    s_mu[WIN], s_rs[WIN];

    const int tid  = threadIdx.x;
    const int lane = tid & 31;
    const int wave = tid >> 5;
    const int colL = lane & 15;
    const int rb   = (lane >> 4) << 3;
    const int t0   = blockIdx.x * WIN;

    __builtin_prefetch(crop_w + tid * 128, 0, 1);

    // gather: inverse (window-reverse + roll-back) for both H taps
    {
        int i  = tid >> 3;
        int c0 = (tid & 7) << 4;
        int t  = t0 + i;
        int b  = t / Lc, r = t % Lc;
        int z  = r / (Hc * Wc); r %= (Hc * Wc);
        int h  = r / Wc, w = r % Wc;
#pragma unroll
        for (int tap = 0; tap < 2; ++tap) {
            int hp = h + tap;
            int zr = (z + 1) & 7;
            int hr = (hp + 2) % Hp;
            int wr = (w + 2) % Wc;
            int zb = zr >> 1, wz = zr & 1;
            int hb = hr >> 2, wh = hr & 3;
            int wb = wr >> 2, ww = wr & 3;
            int lin = ((((zb * 23 + hb) * 45 + wb) * 2 + wz) * 4 + wh) * 4 + ww;
            const _Float16* src = attn_out + ((size_t)b * VOLP + lin) * Cc + c0;
            int off = i * Cc + c0;
#ifdef HAVE_ASYNC_LOAD
            __builtin_amdgcn_global_load_async_to_lds_b128(
                (AS1 v4i*)(uintptr_t)src,       (AS3 v4i*)(uintptr_t)&aw[tap][off], 0, 0);
            __builtin_amdgcn_global_load_async_to_lds_b128(
                (AS1 v4i*)(uintptr_t)(src + 8), (AS3 v4i*)(uintptr_t)&aw[tap][off + 8], 0, 0);
#else
            const uint4* s4 = (const uint4*)src;
            uint4* d4 = (uint4*)&aw[tap][off];
            d4[0] = s4[0]; d4[1] = s4[1];
#endif
        }
#ifdef HAVE_ASYNC_LOAD
        ASYNC_WAIT();
#endif
    }
    __syncthreads();

    // GEMM: sum over taps of (32x128)@(128x128)
    {
        int n0 = wave * 16;
        v8f a0 = {}, a1 = {};
#pragma unroll
        for (int tap = 0; tap < 2; ++tap)
#pragma unroll
            for (int kt = 0; kt < 4; ++kt) {
                v16h fa0 = ld_frag(aw[tap], 0, Cc, kt * 32);
                v16h fa1 = ld_frag(aw[tap], 16, Cc, kt * 32);
                v16h fb  = ld_frag(crop_w + tap * Cc * Cc, n0, Cc, kt * 32);
                a0 = wmma_f16(fa0, fb, a0);
                a1 = wmma_f16(fa1, fb, a1);
            }
        int col = n0 + colL;
        float bb = crop_b[col];
#pragma unroll
        for (int r = 0; r < 8; ++r) {
            cv[(rb + r) * Cc + col]      = a0[r] + bb;
            cv[(16 + rb + r) * Cc + col] = a1[r] + bb;
        }
    }
    __syncthreads();

    if (tid < WIN) {
        float s = 0.f, ss = 0.f;
        for (int c = 0; c < Cc; ++c) { float v = cv[tid * Cc + c]; s += v; ss += v * v; }
        float mu = s * (1.0f / Cc);
        float var = ss * (1.0f / Cc) - mu * mu;
        s_mu[tid] = mu;
        s_rs[tid] = rsqrtf(var + 1e-5f);
    }
    __syncthreads();

    {
        int i  = tid >> 3;
        int c0 = (tid & 7) << 4;
        size_t base = (size_t)(t0 + i) * Cc + c0;
        float mu = s_mu[i], rs = s_rs[i];
#pragma unroll
        for (int c = 0; c < 16; ++c) {
            float v = (cv[i * Cc + c0 + c] - mu) * rs * g1[c0 + c] + be1[c0 + c];
            x_res[base + c] = x[base + c] + v;
        }
    }
}

// ---------------- kernel 3: SwiGLU MLP + LayerNorm + residual ----------------
__global__ __launch_bounds__(256) void k_mlp(
    const float* __restrict__ x_res,
    const _Float16* __restrict__ wg, const float* __restrict__ bg,
    const _Float16* __restrict__ wx, const float* __restrict__ bx,
    const _Float16* __restrict__ w2, const float* __restrict__ b2,
    const float* __restrict__ g2, const float* __restrict__ be2,
    float* __restrict__ out)
{
    __shared__ __align__(16) _Float16 xh[WIN * Cc];    // 8 KB
    __shared__ __align__(16) _Float16 hh[WIN * HID];   // 20 KB
    __shared__ float    mm[WIN * Cc];                  // 16 KB
    __shared__ float    s_mu[WIN], s_rs[WIN];

    const int tid  = threadIdx.x;
    const int lane = tid & 31;
    const int wave = tid >> 5;
    const int colL = lane & 15;
    const int rb   = (lane >> 4) << 3;
    const int t0   = blockIdx.x * WIN;

    __builtin_prefetch(wg + tid * 160, 0, 1);
    __builtin_prefetch(wx + tid * 160, 0, 1);
    __builtin_prefetch(w2 + tid * 160, 0, 1);

    {
        int i  = tid >> 3;
        int c0 = (tid & 7) << 4;
        const float* src = x_res + (size_t)(t0 + i) * Cc + c0;
#pragma unroll
        for (int c = 0; c < 16; ++c) xh[i * Cc + c0 + c] = (_Float16)src[c];
    }
    __syncthreads();

    // GEMM1: (32x128)@(128x320) twice, fused silu(g)*u epilogue
    for (int nt = wave; nt < HID / 16; nt += 8) {
        int n0 = nt * 16;
        v8f g0 = {}, g1v = {}, u0 = {}, u1 = {};
#pragma unroll
        for (int kt = 0; kt < 4; ++kt) {
            v16h fa0 = ld_frag(xh, 0, Cc, kt * 32);
            v16h fa1 = ld_frag(xh, 16, Cc, kt * 32);
            v16h fbg = ld_frag(wg, n0, Cc, kt * 32);
            v16h fbx = ld_frag(wx, n0, Cc, kt * 32);
            g0 = wmma_f16(fa0, fbg, g0);
            g1v = wmma_f16(fa1, fbg, g1v);
            u0 = wmma_f16(fa0, fbx, u0);
            u1 = wmma_f16(fa1, fbx, u1);
        }
        int col = n0 + colL;
        float bgv = bg[col], bxv = bx[col];
#pragma unroll
        for (int r = 0; r < 8; ++r) {
            float g = g0[r] + bgv, u = u0[r] + bxv;
            hh[(rb + r) * HID + col] = (_Float16)((g / (1.0f + __expf(-g))) * u);
            g = g1v[r] + bgv; u = u1[r] + bxv;
            hh[(16 + rb + r) * HID + col] = (_Float16)((g / (1.0f + __expf(-g))) * u);
        }
    }
    __syncthreads();

    // GEMM2: (32x320)@(320x128)
    {
        int n0 = wave * 16;
        v8f a0 = {}, a1 = {};
#pragma unroll
        for (int kt = 0; kt < 10; ++kt) {
            v16h fa0 = ld_frag(hh, 0, HID, kt * 32);
            v16h fa1 = ld_frag(hh, 16, HID, kt * 32);
            v16h fb  = ld_frag(w2, n0, HID, kt * 32);
            a0 = wmma_f16(fa0, fb, a0);
            a1 = wmma_f16(fa1, fb, a1);
        }
        int col = n0 + colL;
        float bb = b2[col];
#pragma unroll
        for (int r = 0; r < 8; ++r) {
            mm[(rb + r) * Cc + col]      = a0[r] + bb;
            mm[(16 + rb + r) * Cc + col] = a1[r] + bb;
        }
    }
    __syncthreads();

    if (tid < WIN) {
        float s = 0.f, ss = 0.f;
        for (int c = 0; c < Cc; ++c) { float v = mm[tid * Cc + c]; s += v; ss += v * v; }
        float mu = s * (1.0f / Cc);
        float var = ss * (1.0f / Cc) - mu * mu;
        s_mu[tid] = mu;
        s_rs[tid] = rsqrtf(var + 1e-5f);
    }
    __syncthreads();

    {
        int i  = tid >> 3;
        int c0 = (tid & 7) << 4;
        size_t base = (size_t)(t0 + i) * Cc + c0;
        float mu = s_mu[i], rs = s_rs[i];
#pragma unroll
        for (int c = 0; c < 16; ++c) {
            float v = (mm[i * Cc + c0 + c] - mu) * rs * g2[c0 + c] + be2[c0 + c];
            out[base + c] = x_res[base + c] + v;
        }
    }
}

// ---------------- launch ----------------
extern "C" void kernel_launch(void* const* d_in, const int* in_sizes, int n_in,
                              void* d_out, int out_size, void* d_ws, size_t ws_size,
                              hipStream_t stream) {
    (void)in_sizes; (void)n_in; (void)out_size; (void)ws_size;
    const float* x        = (const float*)d_in[0];
    const float* w_qkv    = (const float*)d_in[1];
    const float* b_qkv    = (const float*)d_in[2];
    const float* btab     = (const float*)d_in[3];
    const float* w_proj   = (const float*)d_in[4];
    const float* b_proj   = (const float*)d_in[5];
    const float* gamma1   = (const float*)d_in[6];
    const float* beta1    = (const float*)d_in[7];
    const float* gamma2   = (const float*)d_in[8];
    const float* beta2    = (const float*)d_in[9];
    const float* w_g      = (const float*)d_in[10];
    const float* b_g      = (const float*)d_in[11];
    const float* w_x      = (const float*)d_in[12];
    const float* b_x      = (const float*)d_in[13];
    const float* w2       = (const float*)d_in[14];
    const float* b2       = (const float*)d_in[15];
    const float* crop_w   = (const float*)d_in[16];
    const float* crop_b   = (const float*)d_in[17];
    float* out = (float*)d_out;

    char* ws = (char*)d_ws;
    _Float16* qkv_h  = (_Float16*)(ws + OFF_QKVW);
    _Float16* proj_h = (_Float16*)(ws + OFF_PROJW);
    _Float16* crop_h = (_Float16*)(ws + OFF_CROPW);
    _Float16* wg_h   = (_Float16*)(ws + OFF_WG);
    _Float16* wx_h   = (_Float16*)(ws + OFF_WX);
    _Float16* w2_h   = (_Float16*)(ws + OFF_W2);
    float*    bias_e = (float*)   (ws + OFF_BIAS);
    _Float16* attn_o = (_Float16*)(ws + OFF_ATTN);
    float*    x_res  = (float*)   (ws + OFF_XRES);

    // weight prep
    k_cast_f16<<<(3*Cc*Cc + 255)/256, 256, 0, stream>>>(w_qkv, qkv_h, 3*Cc*Cc);
    k_cast_f16<<<(Cc*Cc + 255)/256, 256, 0, stream>>>(w_proj, proj_h, Cc*Cc);
    k_cast_f16<<<(HID*Cc + 255)/256, 256, 0, stream>>>(w_g, wg_h, HID*Cc);
    k_cast_f16<<<(HID*Cc + 255)/256, 256, 0, stream>>>(w_x, wx_h, HID*Cc);
    k_cast_f16<<<(Cc*HID + 255)/256, 256, 0, stream>>>(w2, w2_h, Cc*HID);
    k_cast_crop<<<(2*Cc*Cc + 255)/256, 256, 0, stream>>>(crop_w, crop_h);
    k_bias_expand<<<(HEADS*WIN*WIN + 255)/256, 256, 0, stream>>>(btab, bias_e);

    // fused windowed attention
    k_attn<<<NWB, 256, 0, stream>>>(x, b_qkv, b_proj, qkv_h, proj_h, bias_e, attn_o);

    // crop conv + LN + residual
    k_conv_ln<<<TOK/WIN, 256, 0, stream>>>(x, attn_o, crop_h, crop_b,
                                           gamma1, beta1, x_res);

    // SwiGLU MLP + LN + residual
    k_mlp<<<TOK/WIN, 256, 0, stream>>>(x_res, wg_h, b_g, wx_h, b_x,
                                       w2_h, b2, gamma2, beta2, out);
}
